// FATM_61375082660462
// MI455X (gfx1250) — matrix operand order repaired and checked by reading, not compiled, not run on hardware
//
#include <hip/hip_runtime.h>

// FATM block for MI455X (gfx1250): wave32 + v_wmma_f32_16x16x32_f16.
// T=4,B=8,C=512,H=W=32,nb=16,bs=32. All LDS frag traffic via b128 ops.

#define T_STEPS 4
#define BATCH   8
#define C_CH    512
#define HW      1024
#define EPSBN   1e-5f
#define PITCH   40   // halves; 80B rows -> 16B-aligned b128 slots

typedef __attribute__((ext_vector_type(16))) _Float16 v16h;
typedef __attribute__((ext_vector_type(8)))  _Float16 v8h;
typedef __attribute__((ext_vector_type(4)))  _Float16 h4;
typedef __attribute__((ext_vector_type(8)))  float    v8f;
typedef __attribute__((ext_vector_type(4)))  float    f4;

// orthonormal Haar 32x32 (matches the reference's Kronecker recursion)
__device__ __forceinline__ float haar_entry(int r, int c) {
  const float inv = 0.17677669529663687f;  // 1/sqrt(32)
  if (r == 0) return inv;
  int p = 31 - __clz(r);
  int q = r - (1 << p);
  int seg = 32 >> p;
  int start = q * seg;
  if (c < start || c >= start + seg) return 0.0f;
  float v = exp2f(0.5f * (float)p) * inv;
  return (c < start + (seg >> 1)) ? v : -v;
}

// K base for 16-bit A fragment (16x32), contiguous k = 16*hi + s
__device__ __forceinline__ int a_kbase(int lane, int j) {
  int kb = 2 * j;
  if (lane >= 16) kb += 16;
  return kb;
}

__device__ __forceinline__ v8f wmma_f16(v16h a, v16h b, v8f c) {
  return __builtin_amdgcn_wmma_f32_16x16x32_f16(false, a, false, b, (short)0, c,
                                                false, false);
}

// A frag from row-major tile: one contiguous 16-half run -> 2x ds_load_b128
__device__ __forceinline__ v16h frag_row(const _Float16* rowbase, int hi) {
  union { v16h v; v8h h[2]; } u;
  u.h[0] = *(const v8h*)(rowbase + hi * 16);
  u.h[1] = *(const v8h*)(rowbase + hi * 16 + 8);
  return u.v;
}

// B frag from column-major tile: one contiguous 16-half run -> 2x ds_load_b128
__device__ __forceinline__ v16h frag_col(const _Float16* colbase, int hi) {
  union { v16h v; v8h h[2]; } u;
  u.h[0] = *(const v8h*)(colbase + hi * 16);
  u.h[1] = *(const v8h*)(colbase + hi * 16 + 8);
  return u.v;
}

__device__ __forceinline__ v8h pack8(v8f a) {
  v8h h;
#pragma unroll
  for (int r = 0; r < 8; ++r) h[r] = (_Float16)a[r];
  return h;
}

// ---------------------------------------------------------------- kernel A
__global__ __launch_bounds__(256) void lif_spike_kernel(
    const float* __restrict__ x, _Float16* __restrict__ xs) {
  int i = (blockIdx.x * blockDim.x + threadIdx.x) * 4;
  const size_t stride = (size_t)BATCH * C_CH * HW;
  f4 v = {0.f, 0.f, 0.f, 0.f};
#pragma unroll
  for (int t = 0; t < T_STEPS; ++t) {
    if (t + 1 < T_STEPS) __builtin_prefetch(&x[(t + 1) * stride + i], 0, 0);
    f4 xt = *(const f4*)&x[t * stride + i];
    h4 sp;
#pragma unroll
    for (int e = 0; e < 4; ++e) {
      v[e] = v[e] + (xt[e] - v[e]) * 0.5f;     // tau = 2
      float s = (v[e] >= 1.0f) ? 1.0f : 0.0f;  // vth = 1
      v[e] = (s > 0.0f) ? 0.0f : v[e];         // hard reset
      sp[e] = (_Float16)s;
    }
    *(h4*)&xs[t * stride + i] = sp;
  }
}

// ---------------------------------------------------------------- kernel B
// Per (b,c): for t: W = X*Q^T (A=LDS row-major, B=gen), Y = Q*W (A=gen,
// B=LDS col-major), fused NegIF + BN0 + LIF (reg state) -> hspk f16.
__global__ __launch_bounds__(128) void haar_fwd_kernel(
    const _Float16* __restrict__ xs, _Float16* __restrict__ hspk,
    const float* __restrict__ bnw, const float* __restrict__ bnb,
    const float* __restrict__ bnm, const float* __restrict__ bnv) {
  __shared__ _Float16 Xs[4][32 * PITCH];
  __shared__ _Float16 Ws[4][32 * PITCH];
  int lane = threadIdx.x & 31;
  int wid = threadIdx.x >> 5;
  int wg = blockIdx.x * 4 + wid;  // 0..B*C-1
  int b = wg / C_CH, c = wg % C_CH;
  int hi = (lane >= 16) ? 1 : 0;
  int l15 = lane & 15;
  float inv0 = bnw[c] * rsqrtf(bnv[c] + EPSBN);
  float bet0 = bnb[c] - bnm[c] * inv0;
  _Float16* X = Xs[wid];
  _Float16* W = Ws[wid];

  // loop-invariant generated Q fragments
  v16h qa[2], qbt[2];
#pragma unroll
  for (int ti = 0; ti < 2; ++ti) {
    int m = ti * 16 + l15;
#pragma unroll
    for (int j = 0; j < 8; ++j) {
      int kb = a_kbase(lane, j);
      qa[ti][2 * j]     = (_Float16)haar_entry(m, kb);      // A(m,k)=Q[m][k]
      qa[ti][2 * j + 1] = (_Float16)haar_entry(m, kb + 1);
    }
  }
#pragma unroll
  for (int tj = 0; tj < 2; ++tj) {
    int col = tj * 16 + l15;
    int ko = hi * 16;
#pragma unroll
    for (int j = 0; j < 8; ++j) {
      qbt[tj][2 * j]     = (_Float16)haar_entry(col, ko + 2 * j);  // Q^T(k,n)
      qbt[tj][2 * j + 1] = (_Float16)haar_entry(col, ko + 2 * j + 1);
    }
  }

  float vng[2][2][8], vlf[2][2][8];
#pragma unroll
  for (int ti = 0; ti < 2; ++ti)
#pragma unroll
    for (int tj = 0; tj < 2; ++tj)
#pragma unroll
      for (int r = 0; r < 8; ++r) { vng[ti][tj][r] = 0.f; vlf[ti][tj][r] = 0.f; }

  for (int t = 0; t < T_STEPS; ++t) {
    const _Float16* src = xs + ((size_t)(t * BATCH + b) * C_CH + c) * HW;
    const v8h* srcv = (const v8h*)src;
#pragma unroll
    for (int j = 0; j < 4; ++j)
      *(v8h*)&X[lane * PITCH + j * 8] = srcv[lane * 4 + j];
    __syncthreads();
    // W = X @ Q^T, stored column-major (b128 stores)
#pragma unroll
    for (int ti = 0; ti < 2; ++ti) {
      v16h a = frag_row(&X[(ti * 16 + l15) * PITCH], hi);
      int row0 = ti * 16 + hi * 8;
#pragma unroll
      for (int tj = 0; tj < 2; ++tj) {
        v8f acc = {};
        acc = wmma_f16(a, qbt[tj], acc);
        int col = tj * 16 + l15;
        *(v8h*)&W[col * PITCH + row0] = pack8(acc);
      }
    }
    __syncthreads();
    // Y = Q @ W, fused NegIF + BN0 + LIF
    _Float16* dst = hspk + ((size_t)(t * BATCH + b) * C_CH + c) * HW;
#pragma unroll
    for (int tj = 0; tj < 2; ++tj) {
      int col = tj * 16 + l15;
      v16h bf = frag_col(&W[col * PITCH], hi);
#pragma unroll
      for (int ti = 0; ti < 2; ++ti) {
        v8f y = {};
        y = wmma_f16(qa[ti], bf, y);
        int row0 = ti * 16 + hi * 8;
#pragma unroll
        for (int r = 0; r < 8; ++r) {
          float v = vng[ti][tj][r] + y[r];
          float s = ((v - 1.0f >= 0.f) ? 1.f : 0.f) -
                    ((-v - 1.0f >= 0.f) ? 1.f : 0.f);
          v -= s;  // soft reset, vth=1
          vng[ti][tj][r] = v;
          float z = inv0 * s + bet0;  // BN0
          float vl = vlf[ti][tj][r];
          vl = vl + (z - vl) * 0.5f;  // LIF
          float sp = (vl >= 1.f) ? 1.f : 0.f;
          vlf[ti][tj][r] = (sp > 0.f) ? 0.f : vl;
          dst[(row0 + r) * 32 + col] = (_Float16)sp;
        }
      }
    }
    __syncthreads();
  }
}

// ---------------------------------------------------------------- kernel C
// Block-diagonal channel mix + BN1. A = pixel-major spike tile (row-major),
// B = transposed weights (col-major). Vector epilogue stores.
__global__ __launch_bounds__(256) void mix_kernel(
    const _Float16* __restrict__ hspk, const float* __restrict__ hwgt,
    _Float16* __restrict__ mixed, const float* __restrict__ bnw,
    const float* __restrict__ bnb, const float* __restrict__ bnm,
    const float* __restrict__ bnv) {
  __shared__ _Float16 spT[512 * PITCH];  // [pixel][d]
  __shared__ _Float16 wtc[32 * PITCH];   // [n][d]
  int blk = blockIdx.x;  // n*32 + bl*2 + ph
  int ph = blk & 1, bl = (blk >> 1) & 15, n = blk >> 5;
  int tid = threadIdx.x;
  for (int ch = tid; ch < 2048; ch += 256) {  // 32 ch x 64 8-pixel chunks
    int d = ch >> 6;
    int p0 = (ch & 63) * 8;
    v8h v = *(const v8h*)&hspk[((size_t)n * C_CH + bl * 32 + d) * HW +
                               ph * 512 + p0];
#pragma unroll
    for (int e = 0; e < 8; ++e) spT[(p0 + e) * PITCH + d] = v[e];
  }
  for (int i = tid; i < 1024; i += 256) {  // wtc[n][d] = hwgt[bl][d][n]
    int d = i >> 5, nn = i & 31;
    wtc[nn * PITCH + d] = (_Float16)hwgt[bl * 1024 + i];
  }
  __syncthreads();
  int lane = tid & 31, wid = tid >> 5;
  int hi = (lane >= 16) ? 1 : 0;
  int l15 = lane & 15;
  v16h bq[2];
  float inv1[2], bet1[2];
#pragma unroll
  for (int nt = 0; nt < 2; ++nt) {
    bq[nt] = frag_col(&wtc[(nt * 16 + l15) * PITCH], hi);
    int cg = bl * 32 + nt * 16 + l15;
    inv1[nt] = bnw[C_CH + cg] * rsqrtf(bnv[C_CH + cg] + EPSBN);
    bet1[nt] = bnb[C_CH + cg] - bnm[C_CH + cg] * inv1[nt];
  }
  for (int mt = wid; mt < 32; mt += 8) {
    int mbase = mt * 16;
    v16h a = frag_row(&spT[(mbase + l15) * PITCH], hi);
#pragma unroll
    for (int nt = 0; nt < 2; ++nt) {
      v8f acc = {};
      acc = wmma_f16(a, bq[nt], acc);
      int cg = bl * 32 + nt * 16 + l15;
      v8h o;
#pragma unroll
      for (int r = 0; r < 8; ++r) o[r] = (_Float16)(inv1[nt] * acc[r] + bet1[nt]);
      *(v8h*)&mixed[((size_t)n * C_CH + cg) * HW + ph * 512 + mbase + hi * 8] = o;
    }
  }
}

// ---------------------------------------------------------------- kernel D
// Per (b,c): for t: W = Z*Q (A=LDS row-major, B=gen), Y = Q^T*W (A=gen,
// B=LDS col-major), fused NegIF + BN2 + identity -> d_out f32.
__global__ __launch_bounds__(128) void haar_inv_kernel(
    const _Float16* __restrict__ mixed, const float* __restrict__ x,
    float* __restrict__ out, const float* __restrict__ bnw,
    const float* __restrict__ bnb, const float* __restrict__ bnm,
    const float* __restrict__ bnv) {
  __shared__ _Float16 Zs[4][32 * PITCH];
  __shared__ _Float16 Ws[4][32 * PITCH];
  int lane = threadIdx.x & 31;
  int wid = threadIdx.x >> 5;
  int wg = blockIdx.x * 4 + wid;
  int b = wg / C_CH, c = wg % C_CH;
  int hi = (lane >= 16) ? 1 : 0;
  int l15 = lane & 15;
  float inv2 = bnw[2 * C_CH + c] * rsqrtf(bnv[2 * C_CH + c] + EPSBN);
  float bet2 = bnb[2 * C_CH + c] - bnm[2 * C_CH + c] * inv2;
  _Float16* Z = Zs[wid];
  _Float16* W = Ws[wid];

  v16h qat[2], qb[2];  // A(m,k)=Q[k][m] (Q^T rows); B(k,n)=Q[k][n]
#pragma unroll
  for (int ti = 0; ti < 2; ++ti) {
    int m = ti * 16 + l15;
#pragma unroll
    for (int j = 0; j < 8; ++j) {
      int kb = a_kbase(lane, j);
      qat[ti][2 * j]     = (_Float16)haar_entry(kb, m);
      qat[ti][2 * j + 1] = (_Float16)haar_entry(kb + 1, m);
    }
  }
#pragma unroll
  for (int tj = 0; tj < 2; ++tj) {
    int col = tj * 16 + l15;
    int ko = hi * 16;
#pragma unroll
    for (int j = 0; j < 8; ++j) {
      qb[tj][2 * j]     = (_Float16)haar_entry(ko + 2 * j, col);
      qb[tj][2 * j + 1] = (_Float16)haar_entry(ko + 2 * j + 1, col);
    }
  }

  float vng[2][2][8];
#pragma unroll
  for (int ti = 0; ti < 2; ++ti)
#pragma unroll
    for (int tj = 0; tj < 2; ++tj)
#pragma unroll
      for (int r = 0; r < 8; ++r) vng[ti][tj][r] = 0.f;

  for (int t = 0; t < T_STEPS; ++t) {
    size_t base = ((size_t)(t * BATCH + b) * C_CH + c) * HW;
    const v8h* srcv = (const v8h*)(mixed + base);
#pragma unroll
    for (int j = 0; j < 4; ++j)
      *(v8h*)&Z[lane * PITCH + j * 8] = srcv[lane * 4 + j];
    __syncthreads();
    // W = Z @ Q, stored column-major
#pragma unroll
    for (int ti = 0; ti < 2; ++ti) {
      v16h a = frag_row(&Z[(ti * 16 + l15) * PITCH], hi);
      int row0 = ti * 16 + hi * 8;
#pragma unroll
      for (int tj = 0; tj < 2; ++tj) {
        v8f acc = {};
        acc = wmma_f16(a, qb[tj], acc);
        int col = tj * 16 + l15;
        *(v8h*)&W[col * PITCH + row0] = pack8(acc);
      }
    }
    __syncthreads();
    // Y = Q^T @ W, fused NegIF + BN2 + identity
#pragma unroll
    for (int tj = 0; tj < 2; ++tj) {
      int col = tj * 16 + l15;
      v16h bf = frag_col(&W[col * PITCH], hi);
#pragma unroll
      for (int ti = 0; ti < 2; ++ti) {
        v8f y = {};
        y = wmma_f16(qat[ti], bf, y);
        int row0 = ti * 16 + hi * 8;
#pragma unroll
        for (int r = 0; r < 8; ++r) {
          float v = vng[ti][tj][r] + y[r];
          float s = ((v - 1.0f >= 0.f) ? 1.f : 0.f) -
                    ((-v - 1.0f >= 0.f) ? 1.f : 0.f);
          v -= s;
          vng[ti][tj][r] = v;
          size_t idx = base + (size_t)(row0 + r) * 32 + col;
          out[idx] = inv2 * s + bet2 + x[idx];  // haar branch + identity
        }
      }
    }
    __syncthreads();
  }
}

// ---------------------------------------------------------------- kernel E
// Conv branch: 1x1 (1 WMMA) + 3x3 (9 tap WMMAs) per tile on xs spikes.
// LDS tile [row][x+1][d] pre-padded in x; weights [tap][n][d].
__global__ __launch_bounds__(256) void conv_kernel(
    const _Float16* __restrict__ xs, const float* __restrict__ w1,
    const float* __restrict__ b1, const float* __restrict__ w2,
    const float* __restrict__ b2, float* __restrict__ out,
    const float* __restrict__ bnw, const float* __restrict__ bnb,
    const float* __restrict__ bnm, const float* __restrict__ bnv) {
  __shared__ _Float16 xsh[10 * 34 * 32];  // [ry][x+1][d]
  __shared__ _Float16 w2c[9 * 32 * 32];   // [tap][n][d]
  __shared__ _Float16 w1c[32 * 32];       // [n][d]
  int blk = blockIdx.x;  // n*64 + bl*4 + rg
  int rg = blk & 3, bl = (blk >> 2) & 15, n = blk >> 6;
  int tid = threadIdx.x;
  int r0 = rg * 8;
  for (int u = tid; u < 1280; u += 256) {  // 10 ry x 32 d x 4 8-px chunks
    int ry = u >> 7, rem = u & 127, d = rem >> 2, x0 = (rem & 3) * 8;
    int y = r0 - 1 + ry;
    v8h v = {};
    if (y >= 0 && y < 32)
      v = *(const v8h*)&xs[(((size_t)n * C_CH + bl * 32 + d) * 32 + y) * 32 + x0];
#pragma unroll
    for (int e = 0; e < 8; ++e) xsh[(ry * 34 + x0 + e + 1) * 32 + d] = v[e];
  }
  for (int u = tid; u < 640; u += 256) {  // zero the x = -1 and x = 32 pads
    int ry = u / 64, rem = u & 63, d = rem & 31, side = rem >> 5;
    xsh[(ry * 34 + (side ? 33 : 0)) * 32 + d] = (_Float16)0.0f;
  }
  for (int i = tid; i < 9216; i += 256) {  // w2c[tap][n][d] = w2[n][d][tap]
    int tap = i >> 10, rem = i & 1023, nn = rem >> 5, d = rem & 31;
    w2c[(tap * 32 + nn) * 32 + d] = (_Float16)w2[(nn * 32 + d) * 9 + tap];
  }
  for (int i = tid; i < 1024; i += 256) w1c[i] = (_Float16)w1[i];  // [n][d]
  __syncthreads();
  int lane = tid & 31, wid = tid >> 5;
  int hi = (lane >= 16) ? 1 : 0;
  int l15 = lane & 15;
  int y = r0 + wid;
  int ry = wid + 1;
  v16h b1f[2];
  float inv3[2], bet3[2], inv4[2], bet4[2], bb1[2], bb2[2];
#pragma unroll
  for (int nt = 0; nt < 2; ++nt) {
    b1f[nt] = frag_col(&w1c[(nt * 16 + l15) * 32], hi);
    int k = nt * 16 + l15;
    int cg = bl * 32 + k;
    inv3[nt] = bnw[3 * C_CH + cg] * rsqrtf(bnv[3 * C_CH + cg] + EPSBN);
    bet3[nt] = bnb[3 * C_CH + cg] - bnm[3 * C_CH + cg] * inv3[nt];
    inv4[nt] = bnw[4 * C_CH + cg] * rsqrtf(bnv[4 * C_CH + cg] + EPSBN);
    bet4[nt] = bnb[4 * C_CH + cg] - bnm[4 * C_CH + cg] * inv4[nt];
    bb1[nt] = b1[k];
    bb2[nt] = b2[k];
  }
#pragma unroll
  for (int mt = 0; mt < 2; ++mt) {
    int xl = mt * 16 + l15;
#pragma unroll
    for (int nt = 0; nt < 2; ++nt) {
      v8f acc1 = {}, acc2 = {};
      {  // 1x1 conv
        v16h a = frag_row(&xsh[(ry * 34 + xl + 1) * 32], hi);
        acc1 = wmma_f16(a, b1f[nt], acc1);
      }
#pragma unroll
      for (int tap = 0; tap < 9; ++tap) {  // 3x3 conv, K=32 per tap
        int dy = tap / 3 - 1, dx = tap % 3 - 1;
        v16h a = frag_row(&xsh[((ry + dy) * 34 + xl + 1 + dx) * 32], hi);
        v16h bf = frag_col(&w2c[(tap * 32 + nt * 16 + l15) * 32], hi);
        acc2 = wmma_f16(a, bf, acc2);
      }
      int cg = bl * 32 + nt * 16 + l15;
      int x0 = mt * 16 + hi * 8;
      size_t base = (((size_t)n * C_CH + cg) * 32 + y) * 32 + x0;
      f4* po = (f4*)(out + base);
      f4 o0 = po[0], o1 = po[1];
#pragma unroll
      for (int r = 0; r < 8; ++r) {
        float val = inv3[nt] * (acc1[r] + bb1[nt]) + bet3[nt] +
                    inv4[nt] * (acc2[r] + bb2[nt]) + bet4[nt];
        if (r < 4) o0[r] += val; else o1[r - 4] += val;
      }
      po[0] = o0;
      po[1] = o1;
    }
  }
}

// ---------------------------------------------------------------- launcher
extern "C" void kernel_launch(void* const* d_in, const int* in_sizes, int n_in,
                              void* d_out, int out_size, void* d_ws,
                              size_t ws_size, hipStream_t stream) {
  const float* x    = (const float*)d_in[0];
  const float* hwgt = (const float*)d_in[1];
  const float* w1   = (const float*)d_in[2];
  const float* b1   = (const float*)d_in[3];
  const float* w2   = (const float*)d_in[4];
  const float* b2   = (const float*)d_in[5];
  const float* bnw  = (const float*)d_in[6];
  const float* bnb  = (const float*)d_in[7];
  const float* bnm  = (const float*)d_in[8];
  const float* bnv  = (const float*)d_in[9];
  float* out = (float*)d_out;

  const size_t NELEM = (size_t)T_STEPS * BATCH * C_CH * HW;  // 16,777,216
  _Float16* xsbuf = (_Float16*)d_ws;
  _Float16* hspk  = xsbuf + NELEM;
  _Float16* mixed = hspk + NELEM;

  lif_spike_kernel<<<(BATCH * C_CH * HW) / 1024, 256, 0, stream>>>(x, xsbuf);
  haar_fwd_kernel<<<(BATCH * C_CH) / 4, 128, 0, stream>>>(xsbuf, hspk, bnw,
                                                          bnb, bnm, bnv);
  mix_kernel<<<32 * 16 * 2, 256, 0, stream>>>(hspk, hwgt, mixed, bnw, bnb, bnm,
                                              bnv);
  haar_inv_kernel<<<(BATCH * C_CH) / 4, 128, 0, stream>>>(mixed, x, out, bnw,
                                                          bnb, bnm, bnv);
  conv_kernel<<<32 * 16 * 4, 256, 0, stream>>>(xsbuf, w1, b1, w2, b2, out, bnw,
                                               bnb, bnm, bnv);
}
